// NonLocalPositionAttention_20925080666261
// MI455X (gfx1250) — matrix operand order repaired and checked by reading, not compiled, not run on hardware
//
#include <hip/hip_runtime.h>

typedef __attribute__((ext_vector_type(16))) __bf16 v16bf;
typedef __attribute__((ext_vector_type(8)))  __bf16 v8bf;
typedef __attribute__((ext_vector_type(8)))  float  v8f;

// ---------------------------------------------------------------------------
// WMMA operand loaders (bf16, 16x16x32), contiguous k order k = 16*(lane/16) + s.
// ---------------------------------------------------------------------------

// A-matrix 16x32 (MxK), source row-major, leading dim `lda` (elements).
// lane l: M = l&15; half-wave selects K blocks {0-15} vs {16-31}.
__device__ inline v16bf wmma_load_a(const __bf16* A, int lda) {
  const int l  = threadIdx.x & 31;
  const int m  = l & 15;
  const int kb = (l >> 4) * 16;
  v16bf r;
#pragma unroll
  for (int v = 0; v < 8; ++v) {
    const int k = kb + 2 * v;
    r[2 * v]     = A[m * lda + k];
    r[2 * v + 1] = A[m * lda + k + 1];
  }
  return r;
}

// B-matrix 32x16 (KxN), source row-major [K][N], leading dim `ldb`.
// lane l: N = l&15; K = (l>>4)*16 + 2v (+1).
__device__ inline v16bf wmma_load_b(const __bf16* B, int ldb) {
  const int l  = threadIdx.x & 31;
  const int n  = l & 15;
  const int kb = (l >> 4) * 16;
  v16bf r;
#pragma unroll
  for (int v = 0; v < 8; ++v) {
    r[2 * v]     = B[(kb + 2 * v) * ldb + n];
    r[2 * v + 1] = B[(kb + 2 * v + 1) * ldb + n];
  }
  return r;
}

// B-matrix 32x16 with TRANSPOSED storage Bt[n][k] (leading dim ldt):
// per lane 16 contiguous bf16 (32 B) -> pairs of b128 loads.
__device__ inline v16bf wmma_load_bt(const __bf16* Bt, size_t ldt) {
  const int l  = threadIdx.x & 31;
  const __bf16* p = Bt + (size_t)(l & 15) * ldt + (l >> 4) * 16;
  v16bf r;
#pragma unroll
  for (int v = 0; v < 16; ++v) r[v] = p[v];
  return r;
}

__device__ inline v8f wmma_bf16(v16bf a, v16bf b, v8f c) {
  return __builtin_amdgcn_wmma_f32_16x16x32_bf16(false, a, false, b, (short)0, c,
                                                 false, false);
}

// ---------------------------------------------------------------------------
// Kernel 0: bulk f32 -> bf16 convert of x (done once; the K=4096 attention
// GEMM then streams bf16 out of L2 at half the bytes, zero cvt VALU).
// ---------------------------------------------------------------------------
__global__ __launch_bounds__(256) void convert_bf16_kernel(
    const float* __restrict__ X, __bf16* __restrict__ Xb, size_t total8) {
  size_t i = (size_t)blockIdx.x * blockDim.x + threadIdx.x;
  if (i >= total8) return;
  const float4* p = reinterpret_cast<const float4*>(X) + i * 2;
  float4 a = p[0], b = p[1];
  v8bf o;
  o[0] = (__bf16)a.x; o[1] = (__bf16)a.y; o[2] = (__bf16)a.z; o[3] = (__bf16)a.w;
  o[4] = (__bf16)b.x; o[5] = (__bf16)b.y; o[6] = (__bf16)b.z; o[7] = (__bf16)b.w;
  *reinterpret_cast<v8bf*>(Xb + i * 8) = o;
}

// ---------------------------------------------------------------------------
// Kernel 1: Out = relu(W[M,K] * Xb[b,K,HW] + bias), bf16 WMMA, f32 accum.
// Double-buffered LDS staging: one barrier per K-step, global loads for
// step s+1 overlap the WMMAs of step s.
// Output addressing generic: Out[b*out_bs + m*out_sm + p*out_sp].
//  - e1/e2: OutT=__bf16, out_sm=1, out_sp=D  (transposed [HW][D] layout)
//  - assembly: OutT=float, out_sm=HW, out_sp=1 ([C][HW], straight to d_out)
// ---------------------------------------------------------------------------
template <typename OutT>
__global__ __launch_bounds__(256) void conv1x1_relu_kernel(
    const float* __restrict__ W, const float* __restrict__ bias,
    const __bf16* __restrict__ Xb, OutT* __restrict__ Out, int M, int K,
    int HW, int out_sm, int out_sp, size_t out_bs) {
  // strides are multiples of 8 elements (16 B) so staging stores are b128;
  // 80 B / 144 B row pitches keep the 16 A/B lanes on distinct banks.
  __shared__ __align__(16) __bf16 sW[2][64][40];
  __shared__ __align__(16) __bf16 sX[2][32][72];
  const int tid  = threadIdx.x;
  const int wave = tid >> 5;
  const int n0 = blockIdx.x * 64;
  const int m0 = blockIdx.y * 64;
  const int b  = blockIdx.z;
  const size_t xOff = (size_t)b * K * HW;

  const int wm    = wave >> 1;        // 0..3 : M sub-tile
  const int wnOff = (wave & 1) * 32;  // 0/32 : N sub-pair

  // per-thread staging coordinates (8 contiguous elements each)
  const int wMM = tid >> 2, wKK = (tid & 3) * 8;   // W tile 64x32
  const int xKK = tid >> 3, xNN = (tid & 7) * 8;   // X tile 32x64

  auto loadW = [&](int s) -> v8bf {
    const float4* wp = reinterpret_cast<const float4*>(
        W + (size_t)(m0 + wMM) * K + s * 32 + wKK);
    float4 a = wp[0], c = wp[1];
    v8bf o;
    o[0] = (__bf16)a.x; o[1] = (__bf16)a.y; o[2] = (__bf16)a.z; o[3] = (__bf16)a.w;
    o[4] = (__bf16)c.x; o[5] = (__bf16)c.y; o[6] = (__bf16)c.z; o[7] = (__bf16)c.w;
    return o;
  };
  auto loadX = [&](int s) -> v8bf {
    return *reinterpret_cast<const v8bf*>(
        Xb + xOff + (size_t)(s * 32 + xKK) * HW + n0 + xNN);
  };

  v8f acc0 = {}, acc1 = {};
  const int nsteps = K / 32;

  // prologue: stage step 0 into buffer 0
  {
    v8bf w0 = loadW(0), x0 = loadX(0);
    *reinterpret_cast<v8bf*>(&sW[0][wMM][wKK]) = w0;
    *reinterpret_cast<v8bf*>(&sX[0][xKK][xNN]) = x0;
  }
  __syncthreads();

  for (int s = 0; s < nsteps; ++s) {
    const int cur = s & 1;
    v8bf wnx, xnx;
    const bool more = (s + 1 < nsteps);
    if (more) { wnx = loadW(s + 1); xnx = loadX(s + 1); }

    v16bf a   = wmma_load_a(&sW[cur][wm * 16][0], 40);
    v16bf bb0 = wmma_load_b(&sX[cur][0][wnOff], 72);
    v16bf bb1 = wmma_load_b(&sX[cur][0][wnOff + 16], 72);
    acc0 = wmma_bf16(a, bb0, acc0);
    acc1 = wmma_bf16(a, bb1, acc1);

    if (more) {
      *reinterpret_cast<v8bf*>(&sW[1 - cur][wMM][wKK]) = wnx;
      *reinterpret_cast<v8bf*>(&sX[1 - cur][xKK][xNN]) = xnx;
    }
    __syncthreads();
  }

  const int l = tid & 31;
  const int col = l & 15;
  const int rbase = (l >> 4) * 8;
  OutT* outB = Out + (size_t)b * out_bs;
#pragma unroll
  for (int v = 0; v < 8; ++v) {
    const int mg = m0 + wm * 16 + rbase + v;
    float v0 = acc0[v] + bias[mg];
    float v1 = acc1[v] + bias[mg];
    v0 = v0 > 0.f ? v0 : 0.f;
    v1 = v1 > 0.f ? v1 : 0.f;
    const int p0 = n0 + wnOff + col;
    outB[(size_t)mg * out_sm + (size_t)p0 * out_sp]        = (OutT)v0;
    outB[(size_t)mg * out_sm + (size_t)(p0 + 16) * out_sp] = (OutT)v1;
  }
}

// ---------------------------------------------------------------------------
// 16x16 score tile over K=D=256 with TWO independent accumulator chains
// (even/odd K-chunks). Distinct chains -> distinct B register blocks, so the
// compiler can keep multiple B chunks in flight instead of wait-0 draining
// before every WMMA.
// ---------------------------------------------------------------------------
__device__ inline v8f score_tile(const v16bf* areg, const __bf16* bB, int D) {
  v8f accA = {}, accB = {};
#pragma unroll
  for (int kk = 0; kk < 8; kk += 2) {
    v16bf b0 = wmma_load_bt(bB + kk * 32, D);
    v16bf b1 = wmma_load_bt(bB + (kk + 1) * 32, D);
    accA = wmma_bf16(areg[kk], b0, accA);
    accB = wmma_bf16(areg[kk + 1], b1, accB);
  }
#pragma unroll
  for (int v = 0; v < 8; ++v) accA[v] += accB[v];
  return accA;
}

// ---------------------------------------------------------------------------
// Kernel 2: softmax row stats. scores[i,j] = sum_d e1t[i,d]*e2t[j,d] (K=256).
// Per workgroup: one 16-row i-block; 8 waves stream disjoint j-tiles with
// online max/sum, then shuffle + LDS reduce. Writes m_i and 1/l_i.
// ---------------------------------------------------------------------------
__global__ __launch_bounds__(256) void attn_stats_kernel(
    const __bf16* __restrict__ E1, const __bf16* __restrict__ E2,
    float* __restrict__ statM, float* __restrict__ statIL, int D, int HW) {
  __shared__ float sM[8][16];
  __shared__ float sS[8][16];
  const int tid = threadIdx.x;
  const int wave = tid >> 5;
  const int l = tid & 31;
  const int i0 = blockIdx.x * 16;
  const int b = blockIdx.y;
  const size_t eOff = (size_t)b * HW * D;
  const __bf16* e1 = E1 + eOff + (size_t)i0 * D;
  const __bf16* e2 = E2 + eOff;

  v16bf areg[8];
#pragma unroll
  for (int kk = 0; kk < 8; ++kk) areg[kk] = wmma_load_a(e1 + kk * 32, D);

  float rm[8], rs[8];
#pragma unroll
  for (int v = 0; v < 8; ++v) { rm[v] = -3.0e38f; rs[v] = 0.f; }

  for (int jt = wave; jt < HW / 16; jt += 8) {
    const __bf16* bB = e2 + (size_t)jt * 16 * D;
    __builtin_prefetch(bB + 8 * 16 * D, 0, 1);  // next j-tile for this wave
    v8f acc = score_tile(areg, bB, D);
#pragma unroll
    for (int v = 0; v < 8; ++v) {
      float s = acc[v];
      float m2 = rm[v] > s ? rm[v] : s;
      rs[v] = rs[v] * __expf(rm[v] - m2) + __expf(s - m2);
      rm[v] = m2;
    }
  }
  // reduce the 16 column-lanes of each half-wave
#pragma unroll
  for (int off = 1; off < 16; off <<= 1) {
#pragma unroll
    for (int v = 0; v < 8; ++v) {
      float om = __shfl_xor(rm[v], off, 32);
      float os = __shfl_xor(rs[v], off, 32);
      float m2 = rm[v] > om ? rm[v] : om;
      rs[v] = rs[v] * __expf(rm[v] - m2) + os * __expf(om - m2);
      rm[v] = m2;
    }
  }
  if ((l & 15) == 0) {
    const int rbase = (l >> 4) * 8;
#pragma unroll
    for (int v = 0; v < 8; ++v) {
      sM[wave][rbase + v] = rm[v];
      sS[wave][rbase + v] = rs[v];
    }
  }
  __syncthreads();
  if (tid < 16) {
    float m = sM[0][tid], s = sS[0][tid];
#pragma unroll
    for (int w = 1; w < 8; ++w) {
      float om = sM[w][tid], os = sS[w][tid];
      float m2 = m > om ? m : om;
      s = s * __expf(m - m2) + os * __expf(om - m2);
      m = m2;
    }
    statM[(size_t)b * HW + i0 + tid]  = m;
    statIL[(size_t)b * HW + i0 + tid] = 1.0f / s;
  }
}

// ---------------------------------------------------------------------------
// Kernel 3: per i-block, build P[16][4096] = alpha*softmax row (bf16 in LDS,
// padded to 4104/row so the 16 A-lanes land on distinct banks; ~128 KB, legal
// within CDNA5's 320 KB/workgroup), then d_out += P * xb^T  (K = 4096).
// ---------------------------------------------------------------------------
#define P_LD 4104

__global__ __launch_bounds__(256) void attn_out_kernel(
    const __bf16* __restrict__ E1, const __bf16* __restrict__ E2,
    const __bf16* __restrict__ Xb, const float* __restrict__ statM,
    const float* __restrict__ statIL, const float* __restrict__ alphaPtr,
    float* __restrict__ Out, int C, int D, int HW) {
  __shared__ __align__(16) __bf16 P[16 * P_LD];
  __shared__ float sM[16];
  __shared__ float sIL[16];
  const int tid = threadIdx.x;
  const int wave = tid >> 5;
  const int l = tid & 31;
  const int col = l & 15;
  const int rbase = (l >> 4) * 8;
  const int i0 = blockIdx.x * 16;
  const int b = blockIdx.y;
  const float alpha = alphaPtr[0];
  const size_t eOff = (size_t)b * HW * D;
  const size_t xOff = (size_t)b * (size_t)C * HW;

  if (tid < 16) {
    sM[tid]  = statM[(size_t)b * HW + i0 + tid];
    sIL[tid] = statIL[(size_t)b * HW + i0 + tid];
  }
  __syncthreads();

  const __bf16* e1 = E1 + eOff + (size_t)i0 * D;
  const __bf16* e2 = E2 + eOff;
  v16bf areg[8];
#pragma unroll
  for (int kk = 0; kk < 8; ++kk) areg[kk] = wmma_load_a(e1 + kk * 32, D);

  // Phase 1: scores -> alpha * softmax -> P (bf16) in LDS
  for (int jt = wave; jt < HW / 16; jt += 8) {
    const __bf16* bB = e2 + (size_t)jt * 16 * D;
    __builtin_prefetch(bB + 8 * 16 * D, 0, 1);
    v8f acc = score_tile(areg, bB, D);
#pragma unroll
    for (int v = 0; v < 8; ++v) {
      const int row = rbase + v;
      float p = alpha * __expf(acc[v] - sM[row]) * sIL[row];
      P[row * P_LD + jt * 16 + col] = (__bf16)p;
    }
  }
  __syncthreads();

  // Phase 2: out[i, c] = sum_j P[i,j] * xb[c,j]; accumulate onto assembly
  // term. Two accumulator chains over even/odd K-chunks for load/WMMA overlap.
  const int njk = HW / 32;
  for (int ct = wave; ct < C / 16; ct += 8) {
    const int c0 = ct * 16;
    const __bf16* xB = Xb + xOff + (size_t)c0 * HW;
    v8f accA = {}, accB = {};
    for (int jk = 0; jk < njk; jk += 2) {
      __builtin_prefetch(xB + (jk + 4) * 32, 0, 1);
      v16bf b0 = wmma_load_bt(xB + jk * 32, HW);
      v16bf b1 = wmma_load_bt(xB + (jk + 1) * 32, HW);
      v16bf a0 = wmma_load_a(P + jk * 32, P_LD);
      v16bf a1 = wmma_load_a(P + (jk + 1) * 32, P_LD);
      accA = wmma_bf16(a0, b0, accA);
      accB = wmma_bf16(a1, b1, accB);
    }
    // 8 accumulator rows -> 8 consecutive floats in d_out (32 B aligned)
    float* op = Out + ((size_t)b * C + c0 + col) * HW + i0 + rbase;
    float4 t0 = *reinterpret_cast<float4*>(op);
    float4 t1 = *reinterpret_cast<float4*>(op + 4);
    t0.x += accA[0] + accB[0]; t0.y += accA[1] + accB[1];
    t0.z += accA[2] + accB[2]; t0.w += accA[3] + accB[3];
    t1.x += accA[4] + accB[4]; t1.y += accA[5] + accB[5];
    t1.z += accA[6] + accB[6]; t1.w += accA[7] + accB[7];
    *reinterpret_cast<float4*>(op)     = t0;
    *reinterpret_cast<float4*>(op + 4) = t1;
  }
}

// ---------------------------------------------------------------------------
extern "C" void kernel_launch(void* const* d_in, const int* in_sizes, int n_in,
                              void* d_out, int out_size, void* d_ws,
                              size_t ws_size, hipStream_t stream) {
  const float* x     = (const float*)d_in[0];
  const float* w1    = (const float*)d_in[1];
  const float* b1    = (const float*)d_in[2];
  const float* w2    = (const float*)d_in[3];
  const float* b2    = (const float*)d_in[4];
  const float* w3    = (const float*)d_in[5];
  const float* b3    = (const float*)d_in[6];
  const float* alpha = (const float*)d_in[7];
  float* out = (float*)d_out;

  const int NB = 4, C = 2048, D = 256, HW = 4096;

  char* ws = (char*)d_ws;
  __bf16* xb  = (__bf16*)ws;  ws += (size_t)NB * C * HW * sizeof(__bf16);
  __bf16* e1t = (__bf16*)ws;  ws += (size_t)NB * HW * D * sizeof(__bf16);
  __bf16* e2t = (__bf16*)ws;  ws += (size_t)NB * HW * D * sizeof(__bf16);
  float* statM  = (float*)ws; ws += (size_t)NB * HW * sizeof(float);
  float* statIL = (float*)ws;

  dim3 blk(256);

  // x -> bf16 once (feeds all four GEMMs)
  const size_t total8 = (size_t)NB * C * HW / 8;
  convert_bf16_kernel<<<dim3((unsigned)((total8 + 255) / 256)), blk, 0,
                        stream>>>(x, xb, total8);

  // e1, e2: [D x C] x [C x HW] -> bf16 transposed [HW][D]
  conv1x1_relu_kernel<__bf16><<<dim3(HW / 64, D / 64, NB), blk, 0, stream>>>(
      w1, b1, xb, e1t, D, C, HW, 1, D, (size_t)HW * D);
  conv1x1_relu_kernel<__bf16><<<dim3(HW / 64, D / 64, NB), blk, 0, stream>>>(
      w2, b2, xb, e2t, D, C, HW, 1, D, (size_t)HW * D);
  // assembly: [C x C] x [C x HW] -> f32 [C][HW] directly into d_out
  conv1x1_relu_kernel<float><<<dim3(HW / 64, C / 64, NB), blk, 0, stream>>>(
      w3, b3, xb, out, C, C, HW, HW, 1, (size_t)C * HW);
  // softmax row stats (fused score recompute, never materialize 268 MB)
  attn_stats_kernel<<<dim3(HW / 16, NB), blk, 0, stream>>>(e1t, e2t, statM,
                                                           statIL, D, HW);
  // d_out += alpha * softmax(scores) * xb^T
  attn_out_kernel<<<dim3(HW / 16, NB), blk, 0, stream>>>(
      e1t, e2t, xb, statM, statIL, alpha, out, C, D, HW);
}
